// SingleMchAttnBlock_3092376453888
// MI455X (gfx1250) — compile-verified, matrix operand order checked by reading, not run on hardware
//
#include <hip/hip_runtime.h>
#include <hip/hip_bf16.h>

// ---------------------------------------------------------------------------
// Fused GAT-style attention block for gfx1250 (MI455X), wave32 + WMMA + async.
// Algebraic fusion removes the [B,M,M,64] "We" intermediate entirely:
//   v  = W_edge @ a[2D:]          (16)
//   u1 = W @ a[:D], u2 = W @ a[D:2D]   (128 each)
//   e[b,i,j] = leakyrelu(h[b,i]·u1 + h[b,j]·u2 + comp_val[b,i,j,:]·v)
//   attn = softmax(mask(e));  out = attn @ (h @ W)
// Dense GEMMs: v_wmma_f32_16x16x32_f16 (f32 accumulate).
// Wh tile staged to LDS via global_load_async_to_lds_b128 (ASYNCcnt),
// overlapped with the score/softmax phase.
// ---------------------------------------------------------------------------

typedef __attribute__((ext_vector_type(16))) _Float16 v16h;
typedef __attribute__((ext_vector_type(8)))  float    v8f;

#define B_   8
#define M_   256
#define ND_  128   // node_dim
#define ED_  16    // edge_dim
#define D_   64    // out_dim
#define ALPHA_   0.2f
#define NEG_INF_ (-9.0e15f)

// K index for element e (0..15) of a 16-bit WMMA A/B fragment, hi = lane>>4.
// Per ISA 7.12.2: VGPR0..3 cover K 0..7 (lanes 0-15) / 8..15 (lanes 16-31),
// VGPR4..7 cover K 16..23 / 24..31.
__device__ __forceinline__ int frag_k(int e, int hi) {
    return e + 8 * hi + ((e >= 8) ? 8 : 0);
}

// Async DMA of 16 bytes global -> LDS (VGLOBAL encoding, GV mode).
// VDST = LDS byte offset (low 32 bits of a generic shared pointer are the
// LDS offset per the flat-address truncation rule), VADDR = 64-bit address.
__device__ __forceinline__ void async_copy_b128(unsigned lds_byte_off,
                                                const void* gsrc) {
    unsigned long long ga = (unsigned long long)(size_t)gsrc;
    asm volatile("global_load_async_to_lds_b128 %0, %1, off"
                 :: "v"(lds_byte_off), "v"(ga)
                 : "memory");
}
__device__ __forceinline__ void wait_async0() {
    asm volatile("s_wait_asynccnt 0x0" ::: "memory");
}

// ---------------------------------------------------------------------------
// Kernel 1: fold attention vector 'a' into weights. 1 block of 128 threads.
// ---------------------------------------------------------------------------
__global__ void precompute_kernel(const float* __restrict__ W,
                                  const float* __restrict__ We,
                                  const float* __restrict__ a,
                                  float* __restrict__ u1,
                                  float* __restrict__ u2,
                                  float* __restrict__ vv) {
    const int t = threadIdx.x;
    if (t < ND_) {
        float d1 = 0.f, d2 = 0.f;
        for (int k = 0; k < D_; ++k) {
            float w = W[t * D_ + k];
            d1 += w * a[k];
            d2 += w * a[D_ + k];
        }
        u1[t] = d1;
        u2[t] = d2;
    }
    if (t < ED_) {
        float dv = 0.f;
        for (int k = 0; k < D_; ++k) dv += We[t * D_ + k] * a[2 * D_ + k];
        vv[t] = dv;
    }
}

// ---------------------------------------------------------------------------
// Kernel 2: s1[b,i] = h[b,i]·u1, s2[b,i] = h[b,i]·u2. One thread per (b,i).
// ---------------------------------------------------------------------------
__global__ void scores_kernel(const float* __restrict__ h,
                              const float* __restrict__ u1,
                              const float* __restrict__ u2,
                              float* __restrict__ s1,
                              float* __restrict__ s2) {
    const int idx = blockIdx.x * blockDim.x + threadIdx.x;
    if (idx >= B_ * M_) return;
    const float* hr = h + (size_t)idx * ND_;
    float d1 = 0.f, d2 = 0.f;
#pragma unroll 4
    for (int k = 0; k < ND_; ++k) {
        float x = hr[k];
        d1 += x * u1[k];
        d2 += x * u2[k];
    }
    s1[idx] = d1;
    s2[idx] = d2;
}

// ---------------------------------------------------------------------------
// Kernel 3: Wh = h @ W via WMMA. Block = 128 threads = 4 waves.
// Block handles (b, 16-row tile); wave w owns N-tile w (16 cols).
// Output stored as f16 (B operand of the attention GEMM).
// ---------------------------------------------------------------------------
__global__ __launch_bounds__(128) void wh_gemm_kernel(
    const float* __restrict__ h, const float* __restrict__ W,
    _Float16* __restrict__ WhH) {
    const int blk  = blockIdx.x;   // b*16 + row-tile
    const int b    = blk >> 4;
    const int it   = blk & 15;
    const int wave = threadIdx.x >> 5;
    const int lane = threadIdx.x & 31;
    const int m    = lane & 15;    // row (A) / col (B) within tile
    const int hi   = lane >> 4;

    const float* hrow = h + ((size_t)b * M_ + (size_t)it * 16) * ND_;
    v8f acc = {};
    for (int ks = 0; ks < ND_; ks += 32) {
        v16h af, bf;
#pragma unroll
        for (int e = 0; e < 16; ++e) {
            const int k = ks + frag_k(e, hi);
            af[e] = (_Float16)hrow[(size_t)m * ND_ + k];
            bf[e] = (_Float16)W[(size_t)k * D_ + wave * 16 + m];
        }
        acc = __builtin_amdgcn_wmma_f32_16x16x32_f16(
            false, af, false, bf, (short)0, acc, false, false);
    }
#pragma unroll
    for (int r = 0; r < 8; ++r) {
        const int row = it * 16 + r + 8 * hi;
        WhH[((size_t)b * M_ + row) * D_ + wave * 16 + m] = (_Float16)acc[r];
    }
}

// ---------------------------------------------------------------------------
// Kernel 4: fused edge-score + leaky-ReLU + mask + softmax + (attn @ Wh).
// Block = 256 threads (8 waves) per (b, 16-row tile).
//  - phase 0: kick off async DMA of Wh[b] (256x64 f16 = 32KB) into LDS
//  - phase 1/2: scores + softmax in LDS (hides the DMA)
//  - phase 3: s_wait_asynccnt 0; waves 0..3 run 16x16 WMMA chains from LDS
// ---------------------------------------------------------------------------
__global__ __launch_bounds__(256) void attn_kernel(
    const float* __restrict__ comp_val, const int* __restrict__ mch_mask,
    const float* __restrict__ s1, const float* __restrict__ s2,
    const float* __restrict__ vv, const _Float16* __restrict__ WhH,
    float* __restrict__ out) {
    __shared__ _Float16  WhL[M_][D_];      // 32KB: B operand tile (async DMA)
    __shared__ float     sc[16][M_];       // scores -> exp values (16KB)
    __shared__ _Float16  attnH[16][M_];    // softmax probs, f16 (A operand)
    __shared__ float     redmax[16][16];
    __shared__ float     redsum[16][16];
    __shared__ float     rowmax[16], rowsum[16];
    __shared__ float     s1l[16], s2l[M_], vloc[ED_];

    const int blk = blockIdx.x;
    const int b   = blk >> 4;
    const int it  = blk & 15;
    const int tid = threadIdx.x;

    // ---- phase 0: async-stage Wh[b] into LDS (2048 x 16B chunks) ----
    {
        const char* gsrc = (const char*)(WhH + (size_t)b * M_ * D_);
        const unsigned lds0 = (unsigned)(size_t)(void*)&WhL[0][0];
#pragma unroll
        for (int q = 0; q < 8; ++q) {
            const int idx = q * 256 + tid;          // 16-byte chunk id
            async_copy_b128(lds0 + (unsigned)idx * 16u,
                            gsrc + (size_t)idx * 16u);
        }
    }

    s2l[tid] = s2[b * M_ + tid];
    if (tid < 16)  s1l[tid]  = s1[b * M_ + it * 16 + tid];
    if (tid < ED_) vloc[tid] = vv[tid];
    __syncthreads();

    // ---- phase 1: scores. thread = (row 0..15, 16-col chunk 0..15) ----
    const int row = tid >> 4;
    const int c   = tid & 15;
    const int j0  = c * 16;
    const int gi  = it * 16 + row;               // global query row
    const float sr = s1l[row];
    const float* cvbase = comp_val + (((size_t)b * M_ + gi) * M_ + j0) * ED_;
    const int*   mkbase = mch_mask + ((size_t)b * M_ + gi) * M_ + j0;

    float lmax = -3.0e38f;
#pragma unroll 4
    for (int q = 0; q < 16; ++q) {
        const float4* cv = (const float4*)(cvbase + (size_t)q * ED_);
        float dot = 0.f;
#pragma unroll
        for (int p = 0; p < 4; ++p) {
            float4 x = cv[p];
            dot += x.x * vloc[4 * p + 0] + x.y * vloc[4 * p + 1] +
                   x.z * vloc[4 * p + 2] + x.w * vloc[4 * p + 3];
        }
        float e = sr + s2l[j0 + q] + dot;
        e = (e > 0.f) ? e : ALPHA_ * e;           // leaky relu
        e = (mkbase[q] > 0) ? e : NEG_INF_;       // mask
        sc[row][j0 + q] = e;
        lmax = fmaxf(lmax, e);
    }
    redmax[row][c] = lmax;
    __syncthreads();

    if (c == 0) {
        float rm = redmax[row][0];
#pragma unroll
        for (int q = 1; q < 16; ++q) rm = fmaxf(rm, redmax[row][q]);
        rowmax[row] = rm;
    }
    __syncthreads();

    // ---- phase 2: exp + sum + normalize ----
    const float rm = rowmax[row];
    float lsum = 0.f;
#pragma unroll 4
    for (int q = 0; q < 16; ++q) {
        float ex = __expf(sc[row][j0 + q] - rm);
        sc[row][j0 + q] = ex;
        lsum += ex;
    }
    redsum[row][c] = lsum;
    __syncthreads();

    if (c == 0) {
        float rs = 0.f;
#pragma unroll
        for (int q = 0; q < 16; ++q) rs += redsum[row][q];
        rowsum[row] = rs;
    }
    __syncthreads();

    const float inv = 1.0f / rowsum[row];
#pragma unroll 4
    for (int q = 0; q < 16; ++q)
        attnH[row][j0 + q] = (_Float16)(sc[row][j0 + q] * inv);

    // async DMA must have landed before any lane reads WhL
    wait_async0();
    __syncthreads();

    // ---- phase 3: out[16, 64] = attn[16, 256] @ WhL[256, 64] via WMMA ----
    const int wave = tid >> 5;      // wave-uniform branch: EXEC all-1s inside
    const int lane = tid & 31;
    if (wave < 4) {
        const int m  = lane & 15;
        const int hi = lane >> 4;
        v8f acc = {};
        for (int ks = 0; ks < M_; ks += 32) {
            v16h af, bf;
#pragma unroll
            for (int e = 0; e < 16; ++e) {
                const int k = ks + frag_k(e, hi);
                af[e] = attnH[m][k];
                bf[e] = WhL[k][wave * 16 + m];
            }
            acc = __builtin_amdgcn_wmma_f32_16x16x32_f16(
                false, af, false, bf, (short)0, acc, false, false);
        }
#pragma unroll
        for (int r = 0; r < 8; ++r) {
            const int grow = it * 16 + r + 8 * hi;
            out[((size_t)b * M_ + grow) * D_ + wave * 16 + m] = acc[r];
        }
    }
}

// ---------------------------------------------------------------------------
extern "C" void kernel_launch(void* const* d_in, const int* in_sizes, int n_in,
                              void* d_out, int out_size, void* d_ws, size_t ws_size,
                              hipStream_t stream) {
    (void)in_sizes; (void)n_in; (void)out_size; (void)ws_size;
    const float* h        = (const float*)d_in[0];
    const int*   mch_mask = (const int*)  d_in[1];
    const float* comp_val = (const float*)d_in[2];
    const float* W        = (const float*)d_in[3];
    const float* W_edge   = (const float*)d_in[4];
    const float* a        = (const float*)d_in[5];
    float*       out      = (float*)d_out;

    // workspace layout (floats then f16 Wh)
    float* u1 = (float*)d_ws;           // 128
    float* u2 = u1 + ND_;               // 128
    float* vv = u2 + ND_;               // 16
    float* s1 = vv + ED_;               // B*M
    float* s2 = s1 + B_ * M_;           // B*M
    _Float16* WhH = (_Float16*)(s2 + B_ * M_);   // B*M*D halves

    precompute_kernel<<<1, 128, 0, stream>>>(W, W_edge, a, u1, u2, vv);
    scores_kernel<<<(B_ * M_) / 256, 256, 0, stream>>>(h, u1, u2, s1, s2);
    wh_gemm_kernel<<<B_ * (M_ / 16), 128, 0, stream>>>(h, W, WhH);
    attn_kernel<<<B_ * (M_ / 16), 256, 0, stream>>>(comp_val, mch_mask,
                                                    s1, s2, vv, WhH, out);
}
